// SingleAttention_12713103196390
// MI455X (gfx1250) — matrix operand order repaired and checked by reading, not compiled, not run on hardware
//
#include <hip/hip_runtime.h>
#include <hip/hip_bf16.h>
#include <cstdint>

// Fused single-head attention, MI455X (gfx1250), wave32 WMMA.
// out  : [B,H,S,D] fp32   (first 4,194,304 floats of d_out)
// p_attn:[B,H,S,S] fp32   (next 134,217,728 floats of d_out)

typedef __attribute__((ext_vector_type(16))) _Float16 v16h;
typedef __attribute__((ext_vector_type(8)))  _Float16 v8h;
typedef __attribute__((ext_vector_type(8)))  float    v8f;

#define B_   2
#define H_   16
#define S_   2048
#define D_   64
// exp2 domain: scores are computed as s*log2(e) by pre-scaling Q.
#define QSCALE_  0.1803368801111204f      /* (1/sqrt(64)) * log2(e) */
#define MASKED2_ (-1.4426950408889634e9f) /* -1e9 * log2(e) */

#if defined(__has_builtin)
#if __has_builtin(__builtin_amdgcn_sched_barrier)
#define SCHED_FENCE() __builtin_amdgcn_sched_barrier(0)
#endif
#endif
#ifndef SCHED_FENCE
#define SCHED_FENCE() ((void)0)
#endif

__device__ __forceinline__ v16h pack16(v8h a, v8h b) {
  v16h r;
#pragma unroll
  for (int i = 0; i < 8; ++i) { r[i] = a[i]; r[i + 8] = b[i]; }
  return r;
}

extern "C" __global__ __launch_bounds__(256)
void attn_fused_wmma(const float* __restrict__ Q,
                     const float* __restrict__ K,
                     const float* __restrict__ V,
                     const unsigned char* __restrict__ M,   // jnp.bool_ = 1 byte
                     float* __restrict__ Out,
                     float* __restrict__ P) {
  // LDS: K tile [kv][d] f16, V tile transposed [d][kv] f16, per-wave P strips.
  // Rows padded to 72 halves (144B); all vector chunk reads are 16B-aligned.
  __shared__ _Float16 Ksh[64][72];
  __shared__ _Float16 VshT[64][72];
  __shared__ _Float16 Psh[8][16][72];

  const int head  = blockIdx.x >> 4;     // b*H + h   (32 heads)
  const int qt    = blockIdx.x & 15;     // 128-row q tile
  const int wave  = threadIdx.x >> 5;    // 8 waves / WG
  const int lane  = threadIdx.x & 31;
  const int nn    = lane & 15;           // col index in B/C layouts; A row
  const int hi    = lane >> 4;           // half-wave select
  const int qbase = qt * 128 + wave * 16;
  const int batch = head / H_;

  const float* Qh = Q + (size_t)head * S_ * D_;
  const float* Kh = K + (size_t)head * S_ * D_;
  const float* Vh = V + (size_t)head * S_ * D_;
  const unsigned char* Mb = M + (size_t)batch * S_ * S_;
  float* Oh = Out + (size_t)head * S_ * D_;
  float* Ph = P   + (size_t)head * S_ * S_;

  // ---- Q A-fragments (scaled by log2e/sqrt(D)), 16x32 f16 A layout:
  //   lane row m = lane%16 ; chunk c: e(0..7) -> k = 32c + 8*hi + e,
  //                                   e(8..15)-> k = 32c + 16 + 8*hi + (e-8)
  v16h aq[2];
  {
    const float* qrow = Qh + (size_t)(qbase + nn) * D_;
#pragma unroll
    for (int c = 0; c < 2; ++c) {
      v8h p0, p1;
#pragma unroll
      for (int i = 0; i < 8; ++i) {
        p0[i] = (_Float16)(qrow[32 * c + 8 * hi + i] * QSCALE_);
        p1[i] = (_Float16)(qrow[32 * c + 16 + 8 * hi + i] * QSCALE_);
      }
      aq[c] = pack16(p0, p1);
    }
  }

  // Per-LANE running softmax stats (exp2 domain) over the columns this lane
  // owns (col % 16 == nn).  Cross-lane merge happens once after pass 1.
  float lmax[8], lsum[8];
#pragma unroll
  for (int r = 0; r < 8; ++r) { lmax[r] = -3.0e38f; lsum[r] = 0.0f; }

  // Cooperative global->LDS loaders (256 threads, 64x64 fp32 tile -> f16).
  auto loadK = [&](int kv0) {
    const int t = threadIdx.x, row = t >> 2, cb = (t & 3) << 4;
    const float* src = Kh + (size_t)(kv0 + row) * D_ + cb;
#pragma unroll
    for (int i = 0; i < 4; ++i) {
      float4 f = reinterpret_cast<const float4*>(src)[i];
      _Float16* d = &Ksh[row][cb + 4 * i];
      d[0] = (_Float16)f.x; d[1] = (_Float16)f.y;
      d[2] = (_Float16)f.z; d[3] = (_Float16)f.w;
    }
  };
  auto loadV = [&](int kv0) {  // store transposed: VshT[d][kv]
    const int t = threadIdx.x, row = t >> 2, cb = (t & 3) << 4;
    const float* src = Vh + (size_t)(kv0 + row) * D_ + cb;
#pragma unroll
    for (int i = 0; i < 4; ++i) {
      float4 f = reinterpret_cast<const float4*>(src)[i];
      VshT[cb + 4 * i + 0][row] = (_Float16)f.x;
      VshT[cb + 4 * i + 1][row] = (_Float16)f.y;
      VshT[cb + 4 * i + 2][row] = (_Float16)f.z;
      VshT[cb + 4 * i + 3][row] = (_Float16)f.w;
    }
  };

  // B fragment pair for one kv sub-tile of the score GEMM
  // (B frag: e(0..7) -> d = 8*hi + e, e(8..15) -> d = 16 + 8*hi + (e-8), col = lane%16).
  auto loadBK = [&](int sub, v16h& b0, v16h& b1) {
    const _Float16* kr = &Ksh[sub * 16 + nn][0];
    b0 = pack16(*(const v8h*)(kr + 8 * hi),
                *(const v8h*)(kr + 16 + 8 * hi));
    b1 = pack16(*(const v8h*)(kr + 32 + 8 * hi),
                *(const v8h*)(kr + 48 + 8 * hi));
  };

  // ================= Pass 1: per-lane row max / row sum ====================
  // All 8 B fragments are fetched BEFORE a sched_barrier so they stay live in
  // distinct registers; the 8 WMMAs then retire with staggered dscnt waits
  // instead of a full LDS round-trip per WMMA.  (lmax,lsum) is rebased once
  // per 64-wide kv tile.
  for (int j = 0; j < S_; j += 64) {
    __syncthreads();
    loadK(j);
    if (j + 64 < S_) {   // prefetch next K tile (global_prefetch_b8)
      const int t = threadIdx.x, row = t >> 2, cb = (t & 3) << 4;
      __builtin_prefetch(Kh + (size_t)(j + 64 + row) * D_ + cb, 0, 3);
    }
    __syncthreads();

    v16h b[8];
#pragma unroll
    for (int sub = 0; sub < 4; ++sub) loadBK(sub, b[2 * sub], b[2 * sub + 1]);
    SCHED_FENCE();   // keep all 16 ds_load_b128 above every WMMA

    v8f c0 = {}, c1 = {}, c2 = {}, c3 = {};
    c0 = __builtin_amdgcn_wmma_f32_16x16x32_f16(false, aq[0], false, b[0], (short)0, c0, false, false);
    c1 = __builtin_amdgcn_wmma_f32_16x16x32_f16(false, aq[0], false, b[2], (short)0, c1, false, false);
    c2 = __builtin_amdgcn_wmma_f32_16x16x32_f16(false, aq[0], false, b[4], (short)0, c2, false, false);
    c3 = __builtin_amdgcn_wmma_f32_16x16x32_f16(false, aq[0], false, b[6], (short)0, c3, false, false);
    c0 = __builtin_amdgcn_wmma_f32_16x16x32_f16(false, aq[1], false, b[1], (short)0, c0, false, false);
    c1 = __builtin_amdgcn_wmma_f32_16x16x32_f16(false, aq[1], false, b[3], (short)0, c1, false, false);
    c2 = __builtin_amdgcn_wmma_f32_16x16x32_f16(false, aq[1], false, b[5], (short)0, c2, false, false);
    c3 = __builtin_amdgcn_wmma_f32_16x16x32_f16(false, aq[1], false, b[7], (short)0, c3, false, false);

#pragma unroll
    for (int r = 0; r < 8; ++r) {
      const int qrow = qbase + r + 8 * hi;             // C layout: row r+8*hi
      const unsigned char* mrow = Mb + (size_t)qrow * S_ + j + nn;
      float s0 = mrow[ 0] ? MASKED2_ : c0[r];
      float s1 = mrow[16] ? MASKED2_ : c1[r];
      float s2 = mrow[32] ? MASKED2_ : c2[r];
      float s3 = mrow[48] ? MASKED2_ : c3[r];
      float tm = fmaxf(fmaxf(s0, s1), fmaxf(s2, s3));
      float nm = fmaxf(lmax[r], tm);
      float e  = (exp2f(s0 - nm) + exp2f(s1 - nm)) +
                 (exp2f(s2 - nm) + exp2f(s3 - nm));
      lsum[r] = lsum[r] * exp2f(lmax[r] - nm) + e;
      lmax[r] = nm;
    }
  }

  // ---- one cross-lane (max,sum) merge per row: 4 butterfly steps ----------
  float rmax[8], rinv[8];
#pragma unroll
  for (int r = 0; r < 8; ++r) {
    float m = lmax[r], s = lsum[r];
#pragma unroll
    for (int d = 1; d <= 8; d <<= 1) {   // stays inside the 16-lane half
      float om = __shfl_xor(m, d);
      float os = __shfl_xor(s, d);
      float nm = fmaxf(m, om);
      s = s * exp2f(m - nm) + os * exp2f(om - nm);
      m = nm;
    }
    rmax[r] = m;
    rinv[r] = 1.0f / s;
  }

  v8f acc[4];
#pragma unroll
  for (int dt = 0; dt < 4; ++dt) acc[dt] = {};

  // ======= Pass 2: recompute scores, emit normalized p_attn, out = P@V =====
  for (int j = 0; j < S_; j += 64) {
    __syncthreads();
    loadK(j);
    loadV(j);
    if (j + 64 < S_) {   // prefetch next K+V tiles
      const int t = threadIdx.x, row = t >> 2, cb = (t & 3) << 4;
      __builtin_prefetch(Kh + (size_t)(j + 64 + row) * D_ + cb, 0, 3);
      __builtin_prefetch(Vh + (size_t)(j + 64 + row) * D_ + cb, 0, 3);
    }
    __syncthreads();
#pragma unroll
    for (int sub = 0; sub < 4; ++sub) {
      v16h b0, b1;
      loadBK(sub, b0, b1);
      v8f c = {};
      c = __builtin_amdgcn_wmma_f32_16x16x32_f16(false, aq[0], false, b0, (short)0, c, false, false);
      c = __builtin_amdgcn_wmma_f32_16x16x32_f16(false, aq[1], false, b1, (short)0, c, false, false);
      const int kvcol = j + sub * 16 + nn;
#pragma unroll
      for (int r = 0; r < 8; ++r) {
        const int qrow = qbase + r + 8 * hi;
        float sv = Mb[(size_t)qrow * S_ + kvcol] ? MASKED2_ : c[r];
        float p  = exp2f(sv - rmax[r]) * rinv[r];
        // Streaming store: keep p_attn out of L2 so K/V stay resident.
        __builtin_nontemporal_store(p, &Ph[(size_t)qrow * S_ + kvcol]);
        Psh[wave][r + 8 * hi][sub * 16 + nn] = (_Float16)p;
      }
    }
    // P A-fragments from this wave's private strip (same-wave DS is in-order).
    const _Float16* pr = &Psh[wave][nn][0];
    v16h ap0 = pack16(*(const v8h*)(pr + 8 * hi),
                      *(const v8h*)(pr + 16 + 8 * hi));
    v16h ap1 = pack16(*(const v8h*)(pr + 32 + 8 * hi),
                      *(const v8h*)(pr + 48 + 8 * hi));
    // All 8 V B-fragments live across a sched_barrier -> distinct regs,
    // staggered dscnt waits for the 8 PV WMMAs.
    v16h bv[8];
#pragma unroll
    for (int dt = 0; dt < 4; ++dt) {
      const _Float16* vr = &VshT[dt * 16 + nn][0]; // B frag: e<8 -> kv = 32c+8*hi+e, e>=8 -> kv = 32c+16+8*hi+(e-8)
      bv[2 * dt]     = pack16(*(const v8h*)(vr + 8 * hi),
                              *(const v8h*)(vr + 16 + 8 * hi));
      bv[2 * dt + 1] = pack16(*(const v8h*)(vr + 32 + 8 * hi),
                              *(const v8h*)(vr + 48 + 8 * hi));
    }
    SCHED_FENCE();
#pragma unroll
    for (int dt = 0; dt < 4; ++dt) {
      acc[dt] = __builtin_amdgcn_wmma_f32_16x16x32_f16(false, ap0, false, bv[2 * dt],
                                                       (short)0, acc[dt], false, false);
      acc[dt] = __builtin_amdgcn_wmma_f32_16x16x32_f16(false, ap1, false, bv[2 * dt + 1],
                                                       (short)0, acc[dt], false, false);
    }
  }

  // ---- write out tile: C layout row = r + 8*hi, col = dt*16 + nn
#pragma unroll
  for (int dt = 0; dt < 4; ++dt) {
#pragma unroll
    for (int r = 0; r < 8; ++r) {
      Oh[(size_t)(qbase + r + 8 * hi) * D_ + dt * 16 + nn] = acc[dt][r];
    }
  }
}

extern "C" void kernel_launch(void* const* d_in, const int* in_sizes, int n_in,
                              void* d_out, int out_size, void* d_ws, size_t ws_size,
                              hipStream_t stream) {
  const float* Q = (const float*)d_in[0];
  const float* K = (const float*)d_in[1];
  const float* V = (const float*)d_in[2];
  const unsigned char* M = (const unsigned char*)d_in[3];  // jnp.bool_ (1B)
  float* Out = (float*)d_out;
  float* P   = Out + (size_t)B_ * H_ * S_ * D_;            // p_attn after out

  dim3 grid(B_ * H_ * (S_ / 128));   // 512 workgroups
  dim3 block(256);                   // 8 wave32 per WG
  hipLaunchKernelGGL(attn_fused_wmma, grid, block, 0, stream, Q, K, V, M, Out, P);
}
